// TemporalDiscriminator_45045617001111
// MI455X (gfx1250) — matrix operand order repaired and checked by reading, not compiled, not run on hardware
//
#include <hip/hip_runtime.h>
#include <hip/hip_bf16.h>
#include <math.h>

typedef __bf16 bf16_t;
typedef __attribute__((ext_vector_type(16))) __bf16 v16bf;
typedef __attribute__((ext_vector_type(8)))  __bf16 v8bf;
typedef __attribute__((ext_vector_type(8)))  float   v8f;

// ---------- helpers ----------

__device__ __forceinline__ bf16_t f2bf(float f) {
  union { float f; unsigned u; } x; x.f = f;
  unsigned r = (x.u + 0x7FFFu + ((x.u >> 16) & 1u)) >> 16;  // round-nearest-even
  union { unsigned short s; bf16_t b; } y; y.s = (unsigned short)r;
  return y.b;
}

union Frag16 { v16bf v; v8bf h[2]; };

enum { EPI_BIAS = 1, EPI_GELU = 2, EPI_RES = 4 };

// ---------- batched WMMA GEMM:  C = A[M,K] * B[N,K]^T (+bias,+res,GELU) ----------
// A row-major [M,K]; B row-major [N,K] (weights for x@W.T, K-matrix for Q@K^T,
// pre-transposed V for P@V).  Each wave computes a 32(M) x 64(N) tile with
// 8 f32 accumulators; 8 waves/block stack in M -> block tile 256 x 64.
// Batch z decomposes as (zb, zh) with zh = z % inner; pointers offset by strides.
__global__ __launch_bounds__(256)
void wmma_gemm_bt(const bf16_t* __restrict__ A, long long sAb, long long sAh, int lda,
                  const bf16_t* __restrict__ B, long long sBb, long long sBh, int ldb,
                  float* __restrict__ Cf, bf16_t* __restrict__ Cb,
                  long long sCb, long long sCh, int ldc,
                  const float* __restrict__ bias, const float* __restrict__ res,
                  int M, int N, int K, int inner, int flags)
{
  const int z  = blockIdx.z;
  const int zb = z / inner;
  const int zh = z - zb * inner;
  A += (size_t)zb * (size_t)sAb + (size_t)zh * (size_t)sAh;
  B += (size_t)zb * (size_t)sBb + (size_t)zh * (size_t)sBh;
  const size_t coff = (size_t)zb * (size_t)sCb + (size_t)zh * (size_t)sCh;

  const int lane = threadIdx.x & 31;
  const int wave = threadIdx.x >> 5;
  const int row0 = blockIdx.x * 256 + wave * 32;
  const int col0 = blockIdx.y * 64;
  if (row0 >= M) return;                       // no barriers below: safe early-exit
  const int l15 = lane & 15;
  const int hi  = lane >> 4;

  // A fragment (16x32 MxK): lane half 'hi' holds K {8hi..8hi+7} U {16+8hi..23+8hi}
  const bf16_t* arow0 = A + (size_t)(row0 + l15) * (size_t)lda + (size_t)(hi * 8);
  const bf16_t* arow1 = arow0 + (size_t)16 * (size_t)lda;
  // B fragment (32x16 KxN from [N,K] storage): lane half 'hi' holds K {8hi..8hi+7} U {16+8hi..23+8hi}
  const bf16_t* bcol  = B + (size_t)(col0 + l15) * (size_t)ldb + (size_t)(hi * 8);

  v8f acc[8] = {};                             // [m*4 + n]
  for (int k0 = 0; k0 < K; k0 += 32) {
    Frag16 a0, a1;
    a0.h[0] = *(const v8bf*)(arow0 + k0);
    a0.h[1] = *(const v8bf*)(arow0 + k0 + 16);
    a1.h[0] = *(const v8bf*)(arow1 + k0);
    a1.h[1] = *(const v8bf*)(arow1 + k0 + 16);
    Frag16 b[4];
#pragma unroll
    for (int n = 0; n < 4; ++n) {
      const bf16_t* p = bcol + (size_t)(n * 16) * (size_t)ldb + k0;
      b[n].h[0] = *(const v8bf*)p;
      b[n].h[1] = *(const v8bf*)(p + 16);
    }
    if (k0 + 32 < K) {                         // gfx1250 global_prefetch of next A tile
      __builtin_prefetch(arow0 + k0 + 32, 0, 3);
      __builtin_prefetch(arow1 + k0 + 32, 0, 3);
    }
#pragma unroll
    for (int n = 0; n < 4; ++n) {
      acc[n]     = __builtin_amdgcn_wmma_f32_16x16x32_bf16(false, a0.v, false, b[n].v,
                                                           (short)0, acc[n],     false, false);
      acc[4 + n] = __builtin_amdgcn_wmma_f32_16x16x32_bf16(false, a1.v, false, b[n].v,
                                                           (short)0, acc[4 + n], false, false);
    }
  }

#pragma unroll
  for (int m = 0; m < 2; ++m) {
#pragma unroll
    for (int n = 0; n < 4; ++n) {
      const int col = col0 + n * 16 + l15;
      const float bv = (flags & EPI_BIAS) ? bias[col] : 0.0f;
      v8f a = acc[m * 4 + n];
#pragma unroll
      for (int r = 0; r < 8; ++r) {
        const int row = row0 + m * 16 + r + hi * 8;   // C/D: VGPR r -> M=r(+8 hi half)
        float v = a[r] + bv;
        if (flags & EPI_GELU) v = 0.5f * v * (1.0f + erff(v * 0.70710678118654752f));
        const size_t idx = coff + (size_t)row * (size_t)ldc + (size_t)col;
        if (flags & EPI_RES) v += res[idx];
        if (Cf) Cf[idx] = v;
        if (Cb) Cb[idx] = f2bf(v);
      }
    }
  }
}

// ---------- batched transpose: dst[g][d][j] = src[g][j][d] (bf16) ----------
__global__ __launch_bounds__(256)
void transpose_bf16_k(const bf16_t* __restrict__ src, long long sSb, long long sSh, int ldsrc,
                      bf16_t* __restrict__ dst, long long sDb, long long sDh,
                      int rows, int hd, int inner)
{
  const int z  = blockIdx.z;
  const int zb = z / inner;
  const int zh = z - zb * inner;
  src += (size_t)zb * (size_t)sSb + (size_t)zh * (size_t)sSh;
  dst += (size_t)zb * (size_t)sDb + (size_t)zh * (size_t)sDh;
  const int total = rows * hd;
  for (int i = blockIdx.x * blockDim.x + threadIdx.x; i < total;
       i += gridDim.x * blockDim.x) {
    const int j = i / hd;
    const int d = i - j * hd;                  // coalesced read over d
    dst[(size_t)d * (size_t)rows + (size_t)j] = src[(size_t)j * (size_t)ldsrc + (size_t)d];
  }
}

// ---------- LayerNorm over rows of length D ----------
__global__ __launch_bounds__(256)
void layernorm_rows(const float* __restrict__ X, const float* __restrict__ g,
                    const float* __restrict__ b, bf16_t* __restrict__ Yb,
                    float* __restrict__ Yf, int D)
{
  __shared__ float s1[256], s2[256];
  const int row = blockIdx.x, tid = threadIdx.x;
  const float* x = X + (size_t)row * (size_t)D;
  float a = 0.f, q = 0.f;
  for (int j = tid; j < D; j += 256) { float v = x[j]; a += v; q += v * v; }
  s1[tid] = a; s2[tid] = q; __syncthreads();
  for (int st = 128; st > 0; st >>= 1) {
    if (tid < st) { s1[tid] += s1[tid + st]; s2[tid] += s2[tid + st]; }
    __syncthreads();
  }
  const float mu   = s1[0] / (float)D;
  const float var  = s2[0] / (float)D - mu * mu;
  const float rstd = rsqrtf(var + 1e-5f);
  for (int j = tid; j < D; j += 256) {
    const float v = (x[j] - mu) * rstd * g[j] + b[j];
    const size_t idx = (size_t)row * (size_t)D + (size_t)j;
    if (Yb) Yb[idx] = f2bf(v);
    if (Yf) Yf[idx] = v;
  }
}

// ---------- row softmax (optionally causal: valid cols j <= row) ----------
__global__ __launch_bounds__(256)
void softmax_rows(const float* __restrict__ S, bf16_t* __restrict__ P,
                  int ncol, float scale, int causal)
{
  __shared__ float red[256];
  const int row = blockIdx.x, tid = threadIdx.x;
  const int lim = causal ? row : (ncol - 1);
  const float* s = S + (size_t)row * (size_t)ncol;
  bf16_t* p = P + (size_t)row * (size_t)ncol;

  float mx = -3.0e38f;
  for (int j = tid; j <= lim; j += 256) mx = fmaxf(mx, s[j] * scale);
  red[tid] = mx; __syncthreads();
  for (int st = 128; st > 0; st >>= 1) {
    if (tid < st) red[tid] = fmaxf(red[tid], red[tid + st]);
    __syncthreads();
  }
  mx = red[0]; __syncthreads();

  float sum = 0.f;
  for (int j = tid; j <= lim; j += 256) sum += __expf(s[j] * scale - mx);
  red[tid] = sum; __syncthreads();
  for (int st = 128; st > 0; st >>= 1) {
    if (tid < st) red[tid] += red[tid + st];
    __syncthreads();
  }
  const float inv = 1.0f / red[0];
  for (int j = tid; j < ncol; j += 256)
    p[j] = f2bf(j <= lim ? __expf(s[j] * scale - mx) * inv : 0.0f);
}

// ---------- misc elementwise ----------
__global__ void f32_to_bf16_k(const float* __restrict__ in, bf16_t* __restrict__ out, int n)
{
  const int i = blockIdx.x * blockDim.x + threadIdx.x;
  if (i < n) out[i] = f2bf(in[i]);
}

// x[B,T,C,64,64] -> patches[(B*T*64), 192] (c,py,px) order, bf16
__global__ void patchify_k(const float* __restrict__ x, bf16_t* __restrict__ out, int total)
{
  const int i = blockIdx.x * blockDim.x + threadIdx.x;
  if (i >= total) return;
  const int d  = i % 192;
  const int r  = i / 192;
  const int n  = r & 63;
  const int bt = r >> 6;
  const int c  = d >> 6;
  const int py = (d >> 3) & 7;
  const int px = d & 7;
  const int hn = n >> 3, wn = n & 7;
  const size_t src = (((size_t)bt * 3 + (size_t)c) * 64 + (size_t)(hn * 8 + py)) * 64
                   + (size_t)(wn * 8 + px);
  out[i] = f2bf(x[src]);
}

__global__ void add_pos_k(float* __restrict__ tok, const float* __restrict__ sp,
                          const float* __restrict__ tp, int total)
{
  const int i = blockIdx.x * blockDim.x + threadIdx.x;
  if (i >= total) return;
  const int d = i & 511;
  const int r = i >> 9;
  const int n = r & 63;
  const int t = (r >> 6) & 31;
  tok[i] += sp[n * 512 + d] + tp[t * 512 + d];
}

// Y[o,d] = mean over 'group' consecutive rows of X
__global__ __launch_bounds__(256)
void mean_rows_k(const float* __restrict__ X, float* __restrict__ Y, int group, int D)
{
  const int o = blockIdx.x, tid = threadIdx.x;
  for (int d = tid; d < D; d += 256) {
    float s = 0.f;
    const float* p = X + (size_t)o * (size_t)group * (size_t)D + (size_t)d;
    for (int j = 0; j < group; ++j) s += p[(size_t)j * (size_t)D];
    Y[(size_t)o * (size_t)D + d] = s / (float)group;
  }
}

// out[r] = leaky(z_r @ w1.T + b1, 0.2) @ w2.T + b2 ; hidden = 256 = blockDim
__global__ __launch_bounds__(256)
void mlp_head_k(const float* __restrict__ Z, long long zstride,
                const float* __restrict__ w1, const float* __restrict__ b1,
                const float* __restrict__ w2, const float* __restrict__ b2,
                float* __restrict__ out, int D)
{
  __shared__ float red[256];
  const int r = blockIdx.x, j = threadIdx.x;
  const float* z = Z + (size_t)r * (size_t)zstride;
  const float* w = w1 + (size_t)j * (size_t)D;
  float a = b1[j];
  for (int k = 0; k < D; ++k) a += z[k] * w[k];
  a = a > 0.f ? a : 0.2f * a;
  red[j] = a * w2[j];
  __syncthreads();
  for (int st = 128; st > 0; st >>= 1) { if (j < st) red[j] += red[j + st]; __syncthreads(); }
  if (j == 0) out[r] = red[0] + b2[0];
}

__global__ void finalize_k(const float* __restrict__ t, const float* __restrict__ s,
                           const float* __restrict__ pf, float* __restrict__ out,
                           int B, int T)
{
  const int i = threadIdx.x;
  if (i < B) { out[i] = t[i]; out[B + i] = s[i]; }
  for (int k = i; k < B * T; k += blockDim.x) out[2 * B + k] = pf[k];
  if (i < B) {
    float m = 0.f;
    for (int k = 0; k < T; ++k) m += pf[i * T + k];
    out[2 * B + B * T + i] = t[i] + s[i] + m / (float)T;
  }
}

// ---------- host launcher ----------
extern "C" void kernel_launch(void* const* d_in, const int* in_sizes, int n_in,
                              void* d_out, int out_size, void* d_ws, size_t ws_size,
                              hipStream_t stream)
{
  (void)in_sizes; (void)n_in; (void)out_size; (void)ws_size;

  const int Bsz = 2, T = 32, N = 64, L = T * N;                 // L = 2048
  const int ROWS = Bsz * T * N;                                 // 4096
  const int nS = 3, nL = 6;

  const float* x            = (const float*)d_in[0];
  const float* patch_w      = (const float*)d_in[1];
  const float* patch_b      = (const float*)d_in[2];
  const float* tok_ln_g     = (const float*)d_in[3];
  const float* tok_ln_b     = (const float*)d_in[4];
  const float* spatial_pos  = (const float*)d_in[5];
  const float* temporal_pos = (const float*)d_in[6];
  const float* snorm_g      = (const float*)d_in[7];
  const float* snorm_b      = (const float*)d_in[8];
  const float* sa_in_w      = (const float*)d_in[9];
  const float* sa_in_b      = (const float*)d_in[10];
  const float* sa_out_w     = (const float*)d_in[11];
  const float* sa_out_b     = (const float*)d_in[12];
  const float* ln1_g        = (const float*)d_in[13];
  const float* ln1_b        = (const float*)d_in[14];
  const float* kqv_w        = (const float*)d_in[15];
  const float* kqv_b        = (const float*)d_in[16];
  const float* proj_w       = (const float*)d_in[17];
  const float* proj_b       = (const float*)d_in[18];
  const float* ln2_g        = (const float*)d_in[19];
  const float* ln2_b        = (const float*)d_in[20];
  const float* fc_w         = (const float*)d_in[21];
  const float* fc_b         = (const float*)d_in[22];
  const float* cproj_w      = (const float*)d_in[23];
  const float* cproj_b      = (const float*)d_in[24];
  const float* lnf_g        = (const float*)d_in[25];
  const float* lnf_b        = (const float*)d_in[26];
  const float* th_w1 = (const float*)d_in[27]; const float* th_b1 = (const float*)d_in[28];
  const float* th_w2 = (const float*)d_in[29]; const float* th_b2 = (const float*)d_in[30];
  const float* sh_w1 = (const float*)d_in[31]; const float* sh_b1 = (const float*)d_in[32];
  const float* sh_w2 = (const float*)d_in[33]; const float* sh_b2 = (const float*)d_in[34];
  const float* pf_w1 = (const float*)d_in[35]; const float* pf_b1 = (const float*)d_in[36];
  const float* pf_w2 = (const float*)d_in[37]; const float* pf_b2 = (const float*)d_in[38];

  // ---- workspace partition (256B aligned) ----
  char* wsp = (char*)d_ws;
  size_t off = 0;
  auto alloc = [&](size_t bytes) -> void* {
    void* p = wsp + off;
    off = (off + bytes + 255) & ~(size_t)255;
    return p;
  };
  bf16_t* wPatch = (bf16_t*)alloc((size_t)512 * 192 * 2);
  bf16_t* wSaIn  = (bf16_t*)alloc((size_t)nS * 1536 * 512 * 2);
  bf16_t* wSaOut = (bf16_t*)alloc((size_t)nS * 512 * 512 * 2);
  bf16_t* wKqv   = (bf16_t*)alloc((size_t)nL * 1536 * 512 * 2);
  bf16_t* wProj  = (bf16_t*)alloc((size_t)nL * 512 * 512 * 2);
  bf16_t* wFc    = (bf16_t*)alloc((size_t)nL * 2048 * 512 * 2);
  bf16_t* wCproj = (bf16_t*)alloc((size_t)nL * 512 * 2048 * 2);
  bf16_t* patches = (bf16_t*)alloc((size_t)ROWS * 192 * 2);
  float*  sbuf    = (float*) alloc((size_t)ROWS * 512 * 4);   // residual master (tok / hid)
  bf16_t* lnbuf   = (bf16_t*)alloc((size_t)ROWS * 512 * 2);
  bf16_t* qkv     = (bf16_t*)alloc((size_t)ROWS * 1536 * 2);
  bf16_t* yb      = (bf16_t*)alloc((size_t)ROWS * 512 * 2);
  bf16_t* vt      = (bf16_t*)alloc((size_t)2097152 * 2);      // transposed V (4MB)
  float*  hidf    = (float*) alloc((size_t)ROWS * 512 * 4);   // also pre-LN temp
  char*   scr     = (char*)  alloc((size_t)25165824);         // union scratch (24MB)
  float*  tf      = (float*) alloc((size_t)64 * 512 * 4);
  float*  smean   = (float*) alloc((size_t)2 * 512 * 4);
  float*  tsc     = (float*) alloc(256);
  float*  ssc     = (float*) alloc(256);
  float*  pfb     = (float*) alloc(256 * 4);

  float*  sS  = (float*) scr;                 // spatial scores [256][64][64]
  bf16_t* sP  = (bf16_t*)(scr + 4194304);     // spatial probs
  float*  tS  = (float*) scr;                 // temporal scores [2048][2048]
  bf16_t* tP  = (bf16_t*)(scr + 16777216);    // temporal probs
  bf16_t* mlp = (bf16_t*)scr;                 // fc output [4096][2048]

  auto cvt = [&](const float* src, bf16_t* dst, int n) {
    f32_to_bf16_k<<<(n + 255) / 256, 256, 0, stream>>>(src, dst, n);
  };
  auto gemm = [&](const bf16_t* A, long long sAb, long long sAh, int lda,
                  const bf16_t* Bm, long long sBb, long long sBh, int ldb,
                  float* Cf, bf16_t* Cb, long long sCb, long long sCh, int ldc,
                  const float* bias, const float* res,
                  int M, int Nn, int K, int batches, int inner, int flags) {
    dim3 g((M + 255) / 256, (Nn + 63) / 64, batches);
    wmma_gemm_bt<<<g, 256, 0, stream>>>(A, sAb, sAh, lda, Bm, sBb, sBh, ldb,
                                        Cf, Cb, sCb, sCh, ldc, bias, res,
                                        M, Nn, K, inner, flags);
  };

  // ---- weight conversion to bf16 ----
  cvt(patch_w, wPatch, 512 * 192);
  cvt(sa_in_w,  wSaIn,  nS * 1536 * 512);
  cvt(sa_out_w, wSaOut, nS * 512 * 512);
  cvt(kqv_w,  wKqv,   nL * 1536 * 512);
  cvt(proj_w, wProj,  nL * 512 * 512);
  cvt(fc_w,   wFc,    nL * 2048 * 512);
  cvt(cproj_w, wCproj, nL * 512 * 2048);

  // ---- patchify + patch embed + tok LN + pos ----
  patchify_k<<<(ROWS * 192 + 255) / 256, 256, 0, stream>>>(x, patches, ROWS * 192);
  gemm(patches, 0, 0, 192, wPatch, 0, 0, 192,
       hidf, nullptr, 0, 0, 512, patch_b, nullptr, ROWS, 512, 192, 1, 1, EPI_BIAS);
  layernorm_rows<<<ROWS, 256, 0, stream>>>(hidf, tok_ln_g, tok_ln_b, nullptr, sbuf, 512);
  add_pos_k<<<(ROWS * 512 + 255) / 256, 256, 0, stream>>>(sbuf, spatial_pos, temporal_pos,
                                                          ROWS * 512);

  // ---- spatial attention layers (per frame, nh=4, hd=128, q|k|v order) ----
  for (int i = 0; i < nS; ++i) {
    layernorm_rows<<<ROWS, 256, 0, stream>>>(sbuf, snorm_g, snorm_b, lnbuf, nullptr, 512);
    gemm(lnbuf, 0, 0, 512, wSaIn + (size_t)i * 1536 * 512, 0, 0, 512,
         nullptr, qkv, 0, 0, 1536, sa_in_b + i * 1536, nullptr,
         ROWS, 1536, 512, 1, 1, EPI_BIAS);
    // S = Q @ K^T per (frame, head): 256 batches
    gemm(qkv /*q*/, 98304, 128, 1536, qkv + 512 /*k*/, 98304, 128, 1536,
         sS, nullptr, 16384, 4096, 64, nullptr, nullptr,
         64, 64, 128, 256, 4, 0);
    softmax_rows<<<256 * 64, 256, 0, stream>>>(sS, sP, 64, 0.0883883476483184f, 0);
    // Vt[frame][h][d][j] = V[frame][j][h*128+d]
    transpose_bf16_k<<<dim3(32, 1, 256), 256, 0, stream>>>(
        qkv + 1024, 98304, 128, 1536, vt, 32768, 8192, 64, 128, 4);
    // Y = P @ Vt^T
    gemm(sP, 16384, 4096, 64, vt, 32768, 8192, 64,
         nullptr, yb, 32768, 128, 512, nullptr, nullptr,
         64, 128, 64, 256, 4, 0);
    gemm(yb, 0, 0, 512, wSaOut + (size_t)i * 512 * 512, 0, 0, 512,
         sbuf, nullptr, 0, 0, 512, sa_out_b + i * 512, sbuf,
         ROWS, 512, 512, 1, 1, EPI_BIAS | EPI_RES);
  }

  // ---- temporal transformer layers (causal, nh=8, hd=64, torch split order k|q|v) ----
  for (int i = 0; i < nL; ++i) {
    layernorm_rows<<<ROWS, 256, 0, stream>>>(sbuf, ln1_g + i * 512, ln1_b + i * 512,
                                             lnbuf, nullptr, 512);
    gemm(lnbuf, 0, 0, 512, wKqv + (size_t)i * 1536 * 512, 0, 0, 512,
         nullptr, qkv, 0, 0, 1536, kqv_b + i * 1536, nullptr,
         ROWS, 1536, 512, 1, 1, EPI_BIAS);
    // Vt[b][h][d][j] = V[b][j][h*64+d]  (16 batches, inner=8)
    transpose_bf16_k<<<dim3(512, 1, 16), 256, 0, stream>>>(
        qkv + 1024, (long long)L * 1536, 64, 1536,
        vt, (long long)8 * 64 * L, (long long)64 * L, L, 64, 8);
    for (int b = 0; b < Bsz; ++b) {
      for (int h = 0; h < 8; ++h) {
        const bf16_t* qb  = qkv + (size_t)b * L * 1536 + 512 + h * 64;   // q chunk
        const bf16_t* kb  = qkv + (size_t)b * L * 1536 + 0   + h * 64;   // k chunk
        const bf16_t* vtb = vt + ((size_t)b * 8 + h) * 64 * L;           // [64][L]
        gemm(qb, 0, 0, 1536, kb, 0, 0, 1536,
             tS, nullptr, 0, 0, L, nullptr, nullptr, L, L, 64, 1, 1, 0);
        softmax_rows<<<L, 256, 0, stream>>>(tS, tP, L, 0.125f, 1);
        gemm(tP, 0, 0, L, vtb, 0, 0, L,
             nullptr, yb + (size_t)b * L * 512 + h * 64, 0, 0, 512,
             nullptr, nullptr, L, 64, L, 1, 1, 0);
      }
    }
    gemm(yb, 0, 0, 512, wProj + (size_t)i * 512 * 512, 0, 0, 512,
         sbuf, nullptr, 0, 0, 512, proj_b + i * 512, sbuf,
         ROWS, 512, 512, 1, 1, EPI_BIAS | EPI_RES);
    layernorm_rows<<<ROWS, 256, 0, stream>>>(sbuf, ln2_g + i * 512, ln2_b + i * 512,
                                             lnbuf, nullptr, 512);
    gemm(lnbuf, 0, 0, 512, wFc + (size_t)i * 2048 * 512, 0, 0, 512,
         nullptr, mlp, 0, 0, 2048, fc_b + i * 2048, nullptr,
         ROWS, 2048, 512, 1, 1, EPI_BIAS | EPI_GELU);
    gemm(mlp, 0, 0, 2048, wCproj + (size_t)i * 512 * 2048, 0, 0, 2048,
         sbuf, nullptr, 0, 0, 512, cproj_b + i * 512, sbuf,
         ROWS, 512, 2048, 1, 1, EPI_BIAS | EPI_RES);
  }

  // ---- final LN, means, heads ----
  layernorm_rows<<<ROWS, 256, 0, stream>>>(sbuf, lnf_g, lnf_b, nullptr, hidf, 512);
  mean_rows_k<<<Bsz * T, 256, 0, stream>>>(hidf, tf, N, 512);         // tf [B*T,512]
  mean_rows_k<<<Bsz, 256, 0, stream>>>(hidf, smean, T * N, 512);      // smean [B,512]

  mlp_head_k<<<Bsz, 256, 0, stream>>>(tf + (size_t)(T - 1) * 512, (long long)T * 512,
                                      th_w1, th_b1, th_w2, th_b2, tsc, 512);
  mlp_head_k<<<Bsz, 256, 0, stream>>>(smean, 512, sh_w1, sh_b1, sh_w2, sh_b2, ssc, 512);
  mlp_head_k<<<Bsz * T, 256, 0, stream>>>(tf, 512, pf_w1, pf_b1, pf_w2, pf_b2, pfb, 512);

  finalize_k<<<1, 128, 0, stream>>>(tsc, ssc, pfb, (float*)d_out, Bsz, T);
}
